// DeformCrossTransLayer_9208409883242
// MI455X (gfx1250) — compile-verified, matrix operand order checked by reading, not run on hardware
//
#include <hip/hip_runtime.h>

typedef __attribute__((ext_vector_type(16))) _Float16 v16h;
typedef __attribute__((ext_vector_type(8)))  _Float16 v8h;
typedef __attribute__((ext_vector_type(8)))  float    v8f;

// Problem constants (from reference setup_inputs)
// B=2, N=1024, C=128, GROUPS=HEADS=8, DIM_HEAD=64, INNER=512, M=256, HID=32

// ---------------------------------------------------------------------------
// Fragment loader for row-major f16 matrices (global or LDS source).
// 16-bit A (16x32) / B (32x16) WMMA layout (CDNA5 ISA 7.12.2):
//   lane<16 : halves 0..7 = K0..7,   halves 8..15 = K16..23
//   lane>=16: halves 0..7 = K8..15,  halves 8..15 = K24..31
// ---------------------------------------------------------------------------
__device__ inline v16h load_frag(const _Float16* rowbase, int k0, int lane) {
  union { v16h v; v8h h[2]; } u;
  int off = (lane < 16) ? 0 : 8;
  u.h[0] = *(const v8h*)(rowbase + k0 + off);
  u.h[1] = *(const v8h*)(rowbase + k0 + off + 16);
  return u.v;
}

// Async-copy 16 bytes global -> LDS (GLOBAL_LOAD_ASYNC_TO_LDS_B128, ASYNCcnt).
__device__ inline void async_copy_b128(const void* gptr, void* lds_ptr) {
  unsigned long long ga = (unsigned long long)gptr;
  unsigned ldsoff = (unsigned)(uintptr_t)lds_ptr;   // low 32 bits = LDS byte addr
  asm volatile("global_load_async_to_lds_b128 %0, %1, off"
               :: "v"(ldsoff), "v"(ga) : "memory");
}
__device__ inline void async_wait0() {
  asm volatile("s_wait_asynccnt 0x0" ::: "memory");
}

// ---------------------------------------------------------------------------
// 1) LayerNorm of x1 and x2, output transposed to (B, C, N) fp32
// ---------------------------------------------------------------------------
__global__ void k_layernorm(const float* __restrict__ x1, const float* __restrict__ x2,
                            const float* __restrict__ gam, const float* __restrict__ bet,
                            float* __restrict__ n1, float* __restrict__ n2) {
  int bn = blockIdx.x;              // 0..2047
  int b = bn >> 10, n = bn & 1023;
  int c = threadIdx.x;              // 0..127
  __shared__ float red[128];
  for (int which = 0; which < 2; ++which) {
    const float* x = which ? x2 : x1;
    float* o = which ? n2 : n1;
    float v = x[(b * 1024 + n) * 128 + c];
    red[c] = v; __syncthreads();
    for (int s = 64; s > 0; s >>= 1) { if (c < s) red[c] += red[c + s]; __syncthreads(); }
    float mean = red[0] * (1.0f / 128.0f);
    __syncthreads();
    float d = v - mean;
    red[c] = d * d; __syncthreads();
    for (int s = 64; s > 0; s >>= 1) { if (c < s) red[c] += red[c + s]; __syncthreads(); }
    float var = red[0] * (1.0f / 128.0f);
    __syncthreads();
    o[(b * 128 + c) * 1024 + n] = d * rsqrtf(var + 1e-5f) * gam[c] + bet[c];
  }
}

// ---------------------------------------------------------------------------
// 2) grouped q projection: q (B,512,N) fp32 + scaled f16 q^T tiles
//    qA layout: [b][h][n][d] row-major (A fragments for sim GEMM)
// ---------------------------------------------------------------------------
__global__ void k_qproj(const float* __restrict__ n1, const float* __restrict__ wq,
                        float* __restrict__ qf, _Float16* __restrict__ qA) {
  int idx = blockIdx.x * blockDim.x + threadIdx.x;   // 2*512*1024 threads
  int n = idx & 1023;
  int o = (idx >> 10) & 511;
  int b = idx >> 19;
  int g = o >> 6;
  float acc = 0.f;
  const float* w  = wq + o * 16;
  const float* xp = n1 + (b * 128 + g * 16) * 1024 + n;
#pragma unroll
  for (int d = 0; d < 16; ++d) acc += w[d] * xp[d * 1024];
  qf[(b * 512 + o) * 1024 + n] = acc;
  int h = g, dd = o & 63;
  qA[(((b * 8 + h) * 1024) + n) * 64 + dd] = (_Float16)(acc * 0.125f); // *DIM_HEAD^-0.5
}

// ---------------------------------------------------------------------------
// 3) offset network: depthwise conv k=6 s=4 pad=1 -> GELU -> 1x1 -> tanh*2
//    outputs vgrid_scaled (BG, M)
// ---------------------------------------------------------------------------
__global__ void k_offsets(const float* __restrict__ qf, const float* __restrict__ w1,
                          const float* __restrict__ b1, const float* __restrict__ w2,
                          float* __restrict__ vg) {
  int bgm = blockIdx.x;             // 0..4095
  int bg = bgm >> 8, m = bgm & 255;
  int c = threadIdx.x;              // 0..63
  const float* row = qf + (bg * 64 + c) * 1024;
  float acc = 0.f;
#pragma unroll
  for (int t = 0; t < 6; ++t) {
    int pos = 4 * m - 1 + t;
    float xv = (pos >= 0 && pos < 1024) ? row[pos] : 0.f;
    acc += w1[c * 6 + t] * xv;
  }
  acc += b1[c];
  float ge = 0.5f * acc * (1.0f + erff(acc * 0.70710678118654752f)); // exact GELU
  __shared__ float red[64];
  red[c] = ge * w2[c];
  __syncthreads();
  for (int s = 32; s > 0; s >>= 1) { if (c < s) red[c] += red[c + s]; __syncthreads(); }
  if (c == 0) {
    float off = tanhf(red[0]) * 2.0f;
    float vgrid = (float)m + off;
    vg[bg * 256 + m] = 2.0f * vgrid / 255.0f - 1.0f;
  }
}

// ---------------------------------------------------------------------------
// 4) deformable gather of kv + grouped k/v projections
//    kT f16 [b][h][m][d]  (B fragments for sim GEMM)
//    vB f16 [b][h][d][m]  (B fragments for out GEMM)
// ---------------------------------------------------------------------------
__global__ void k_kvproj(const float* __restrict__ n2, const float* __restrict__ vg,
                         const float* __restrict__ wk, const float* __restrict__ wv,
                         _Float16* __restrict__ kT, _Float16* __restrict__ vB) {
  int bgm = blockIdx.x; int bg = bgm >> 8, m = bgm & 255;
  int b = bg >> 3, g = bg & 7;
  int t = threadIdx.x;              // 0..63
  __shared__ float kv[16];
  if (t < 16) {
    float grid = vg[bg * 256 + m];
    float ix = ((grid + 1.0f) * 1024.0f - 1.0f) * 0.5f;
    float i0f = floorf(ix);
    float wlin = ix - i0f;
    int i0 = (int)i0f;
    const float* row = n2 + (b * 128 + g * 16 + t) * 1024;
    float g0 = (i0 >= 0 && i0 < 1024) ? row[i0] : 0.f;
    int i1 = i0 + 1;
    float g1 = (i1 >= 0 && i1 < 1024) ? row[i1] : 0.f;
    kv[t] = g0 * (1.0f - wlin) + g1 * wlin;
  }
  __syncthreads();
  const float* wkr = wk + (g * 64 + t) * 16;
  const float* wvr = wv + (g * 64 + t) * 16;
  float ka = 0.f, va = 0.f;
#pragma unroll
  for (int d = 0; d < 16; ++d) { ka += wkr[d] * kv[d]; va += wvr[d] * kv[d]; }
  kT[(bg * 256 + m) * 64 + t] = (_Float16)ka;
  vB[(bg * 64 + t) * 256 + m] = (_Float16)va;
}

// ---------------------------------------------------------------------------
// 5) sim = q^T k via WMMA. 4 waves/block share one (b,h); the 32 KB k^T panel
//    is staged once per block into LDS with async global->LDS copies, then
//    B fragments are served from LDS (ds_load_b128).
// ---------------------------------------------------------------------------
__global__ void k_sim(const _Float16* __restrict__ qA, const _Float16* __restrict__ kT,
                      float* __restrict__ sim) {
  __shared__ _Float16 kbuf[256 * 64];     // 32 KB panel: all keys for this (b,h)
  int tid = threadIdx.x;
  int lane = tid & 31;
  int bh = blockIdx.x >> 4;               // 16 blocks (64 n-tiles / 4 waves) per bh
  {
    const _Float16* src = kT + bh * (256 * 64);
#pragma unroll
    for (int i = 0; i < 16; ++i) {
      int e = (i * 128 + tid) * 8;        // 16-byte chunk index
      async_copy_b128(src + e, &kbuf[e]);
    }
    async_wait0();
  }
  __syncthreads();

  int w = blockIdx.x * 4 + (tid >> 5);    // 0..1023
  int nt = w & 63;
  int n0 = nt * 16;
  int col = lane & 15;
  int rbase = (lane < 16) ? 0 : 8;
  const _Float16* arow = qA + ((bh * 1024) + n0 + col) * 64;
  v16h a0 = load_frag(arow, 0, lane);
  v16h a1 = load_frag(arow, 32, lane);
  for (int mt = 0; mt < 16; ++mt) {
    int m0 = mt * 16;
    const _Float16* brow = kbuf + (m0 + col) * 64;
    v16h b0 = load_frag(brow, 0, lane);
    v16h b1 = load_frag(brow, 32, lane);
    v8f c = {};
    c = __builtin_amdgcn_wmma_f32_16x16x32_f16(false, a0, false, b0, (short)0, c, false, false);
    c = __builtin_amdgcn_wmma_f32_16x16x32_f16(false, a1, false, b1, (short)0, c, false, false);
    float* sp = sim + ((bh * 1024) + n0 + rbase) * 256 + m0 + col;
#pragma unroll
    for (int r = 0; r < 8; ++r) sp[r * 256] = c[r];
  }
}

// ---------------------------------------------------------------------------
// 6) continuous-position-bias MLP on WMMA, accumulated into sim.
//    One wave per (bg,n); 16 m-tiles of 16 positions.
// ---------------------------------------------------------------------------
__global__ void k_cpb(const float* __restrict__ vg,
                      const float* __restrict__ w0, const float* __restrict__ b0,
                      const float* __restrict__ w1, const float* __restrict__ b1,
                      const float* __restrict__ w2, const float* __restrict__ b2,
                      float* __restrict__ sim) {
  int lane = threadIdx.x & 31;
  int w = blockIdx.x * 8 + (threadIdx.x >> 5);   // 0..16383
  int bg = w >> 10, n = w & 1023;
  int col = lane & 15;
  // constant B fragments of w1^T (B[k][j] = w1[j][k]); two 32x16 tiles
  union { v16h v; _Float16 e[16]; } bw0, bw1;
#pragma unroll
  for (int hh = 0; hh < 8; ++hh) {
    int k  = (lane < 16) ? hh : 8 + hh;
    int k2 = 16 + k;
    bw0.e[hh]     = (_Float16)w1[col * 32 + k];
    bw0.e[8 + hh] = (_Float16)w1[col * 32 + k2];
    bw1.e[hh]     = (_Float16)w1[(16 + col) * 32 + k];
    bw1.e[8 + hh] = (_Float16)w1[(16 + col) * 32 + k2];
  }
  float gq = 2.0f * (float)n / 1023.0f - 1.0f;
  int j0 = col, j1 = 16 + col;
  float w2j0 = w2[j0], w2j1 = w2[j1];
  float b1j0 = b1[j0], b1j1 = b1[j1];
  float bias2 = b2[0];
  for (int mt = 0; mt < 16; ++mt) {
    int m0 = mt * 16;
    float p = gq - vg[bg * 256 + m0 + col];
    float penc = copysignf(log1pf(fabsf(p)), p);
    union { v16h v; _Float16 e[16]; } a;
#pragma unroll
    for (int hh = 0; hh < 8; ++hh) {
      int k  = (lane < 16) ? hh : 8 + hh;
      int k2 = 16 + k;
      float t0 = fmaxf(penc * w0[k]  + b0[k],  0.f);
      float t1 = fmaxf(penc * w0[k2] + b0[k2], 0.f);
      a.e[hh]     = (_Float16)t0;
      a.e[8 + hh] = (_Float16)t1;
    }
    v8f c0 = {}, c1 = {};
    c0 = __builtin_amdgcn_wmma_f32_16x16x32_f16(false, a.v, false, bw0.v, (short)0, c0, false, false);
    c1 = __builtin_amdgcn_wmma_f32_16x16x32_f16(false, a.v, false, bw1.v, (short)0, c1, false, false);
    float acc[8];
#pragma unroll
    for (int r = 0; r < 8; ++r)
      acc[r] = fmaxf(c0[r] + b1j0, 0.f) * w2j0 + fmaxf(c1[r] + b1j1, 0.f) * w2j1;
    // reduce hidden dim across each 16-lane half (halves stay disjoint for mask<16)
    for (int mask = 1; mask < 16; mask <<= 1) {
#pragma unroll
      for (int r = 0; r < 8; ++r) acc[r] += __shfl_xor(acc[r], mask, 32);
    }
    if (col == 0) {
      int base = (lane < 16) ? 0 : 8;   // lanes<16 hold rows 0..7, lanes>=16 rows 8..15
      float* sp = sim + ((bg * 1024) + n) * 256 + m0 + base;
#pragma unroll
      for (int r = 0; r < 8; ++r) sp[r] += acc[r] + bias2;
    }
  }
}

// ---------------------------------------------------------------------------
// 7) softmax over M=256, emit attn f16 [b][h][n][m]
// ---------------------------------------------------------------------------
__global__ void k_softmax(const float* __restrict__ sim, _Float16* __restrict__ attn) {
  int lane = threadIdx.x & 31;
  int row = blockIdx.x * 8 + (threadIdx.x >> 5);  // 0..16383
  const float* sp = sim + (size_t)row * 256;
  float vals[8];
  float mx = -1e30f;
#pragma unroll
  for (int k = 0; k < 8; ++k) { vals[k] = sp[lane + 32 * k]; mx = fmaxf(mx, vals[k]); }
  for (int mask = 1; mask < 32; mask <<= 1) mx = fmaxf(mx, __shfl_xor(mx, mask, 32));
  float sum = 0.f;
#pragma unroll
  for (int k = 0; k < 8; ++k) { vals[k] = expf(vals[k] - mx); sum += vals[k]; }
  for (int mask = 1; mask < 32; mask <<= 1) sum += __shfl_xor(sum, mask, 32);
  float inv = 1.0f / sum;
  _Float16* ap = attn + (size_t)row * 256;
#pragma unroll
  for (int k = 0; k < 8; ++k) ap[lane + 32 * k] = (_Float16)(vals[k] * inv);
}

// ---------------------------------------------------------------------------
// 8) out = attn @ v^T via WMMA. 4 waves/block share one (b,h); the 32 KB v
//    panel is staged once per block into LDS via async copies.
//    outT fp32 [b][n][512]
// ---------------------------------------------------------------------------
__global__ void k_out(const _Float16* __restrict__ attn, const _Float16* __restrict__ vB,
                      float* __restrict__ outT) {
  __shared__ _Float16 vbuf[64 * 256];     // 32 KB panel: all values for this (b,h)
  int tid = threadIdx.x;
  int lane = tid & 31;
  int bh = blockIdx.x >> 4;
  {
    const _Float16* src = vB + bh * (64 * 256);
#pragma unroll
    for (int i = 0; i < 16; ++i) {
      int e = (i * 128 + tid) * 8;
      async_copy_b128(src + e, &vbuf[e]);
    }
    async_wait0();
  }
  __syncthreads();

  int w = blockIdx.x * 4 + (tid >> 5);    // 0..1023
  int nt = w & 63;
  int b = bh >> 3, h = bh & 7;
  int n0 = nt * 16;
  int col = lane & 15;
  int rbase = (lane < 16) ? 0 : 8;
  v8f c[4] = {};
  const _Float16* arow = attn + ((bh * 1024) + n0 + col) * 256;
  for (int kc = 0; kc < 8; ++kc) {
    int k0 = kc * 32;
    __builtin_prefetch(arow + k0 + 32, 0, 0);
    v16h af = load_frag(arow, k0, lane);
#pragma unroll
    for (int dt = 0; dt < 4; ++dt) {
      const _Float16* brow = vbuf + (dt * 16 + col) * 256;
      v16h bf = load_frag(brow, k0, lane);
      c[dt] = __builtin_amdgcn_wmma_f32_16x16x32_f16(false, af, false, bf, (short)0, c[dt], false, false);
    }
  }
#pragma unroll
  for (int dt = 0; dt < 4; ++dt) {
    int cc = h * 64 + dt * 16 + col;
    float* op = outT + ((b * 1024) + n0 + rbase) * 512 + cc;
#pragma unroll
    for (int r = 0; r < 8; ++r) op[r * 512] = c[dt][r];
  }
}

// ---------------------------------------------------------------------------
// 9) output projection + residual: out[b][n][c] = x1 + wo @ outrow + bo
// ---------------------------------------------------------------------------
__global__ void k_final(const float* __restrict__ x1, const float* __restrict__ outT,
                        const float* __restrict__ wo, const float* __restrict__ bo,
                        float* __restrict__ out) {
  int bn = blockIdx.x; int b = bn >> 10, n = bn & 1023;
  int c = threadIdx.x;             // 0..127
  __shared__ float row[512];
  for (int k = c; k < 512; k += 128) row[k] = outT[(b * 1024 + n) * 512 + k];
  __syncthreads();
  float acc = bo[c];
  const float* wr = wo + c * 512;
  for (int k = 0; k < 512; ++k) acc += wr[k] * row[k];
  out[(b * 1024 + n) * 128 + c] = x1[(b * 1024 + n) * 128 + c] + acc;
}

// ---------------------------------------------------------------------------
extern "C" void kernel_launch(void* const* d_in, const int* in_sizes, int n_in,
                              void* d_out, int out_size, void* d_ws, size_t ws_size,
                              hipStream_t stream) {
  const float* x1     = (const float*)d_in[0];
  const float* x2     = (const float*)d_in[1];
  const float* ln_g   = (const float*)d_in[2];
  const float* ln_b   = (const float*)d_in[3];
  const float* wq     = (const float*)d_in[4];
  const float* wk     = (const float*)d_in[5];
  const float* wv     = (const float*)d_in[6];
  const float* wo     = (const float*)d_in[7];
  const float* bo     = (const float*)d_in[8];
  const float* off_w1 = (const float*)d_in[9];
  const float* off_b1 = (const float*)d_in[10];
  const float* off_w2 = (const float*)d_in[11];
  const float* cw0    = (const float*)d_in[12];
  const float* cb0    = (const float*)d_in[13];
  const float* cw1    = (const float*)d_in[14];
  const float* cb1    = (const float*)d_in[15];
  const float* cw2    = (const float*)d_in[16];
  const float* cb2    = (const float*)d_in[17];
  (void)in_sizes; (void)n_in; (void)out_size; (void)ws_size;

  char* ws = (char*)d_ws;
  size_t off = 0;
  auto alloc = [&](size_t bytes) -> void* {
    void* p = ws + off;
    off += (bytes + 255) & ~(size_t)255;
    return p;
  };
  float*    n1   = (float*)   alloc((size_t)2 * 128 * 1024 * 4);  // 1 MB
  float*    n2   = (float*)   alloc((size_t)2 * 128 * 1024 * 4);  // 1 MB
  float*    qf   = (float*)   alloc((size_t)2 * 512 * 1024 * 4);  // 4 MB
  _Float16* qA   = (_Float16*)alloc((size_t)2 * 8 * 1024 * 64 * 2);  // 2 MB
  float*    vg   = (float*)   alloc((size_t)16 * 256 * 4);           // 16 KB
  _Float16* kT   = (_Float16*)alloc((size_t)16 * 256 * 64 * 2);      // 0.5 MB
  _Float16* vB   = (_Float16*)alloc((size_t)16 * 64 * 256 * 2);      // 0.5 MB
  float*    sim  = (float*)   alloc((size_t)2 * 8 * 1024 * 256 * 4); // 16 MB
  _Float16* attn = (_Float16*)alloc((size_t)2 * 8 * 1024 * 256 * 2); // 8 MB
  float*    outT = (float*)   alloc((size_t)2 * 1024 * 512 * 4);     // 4 MB

  k_layernorm<<<dim3(2048), dim3(128), 0, stream>>>(x1, x2, ln_g, ln_b, n1, n2);
  k_qproj    <<<dim3(4096), dim3(256), 0, stream>>>(n1, wq, qf, qA);
  k_offsets  <<<dim3(4096), dim3(64),  0, stream>>>(qf, off_w1, off_b1, off_w2, vg);
  k_kvproj   <<<dim3(4096), dim3(64),  0, stream>>>(n2, vg, wk, wv, kT, vB);
  k_sim      <<<dim3(256),  dim3(128), 0, stream>>>(qA, kT, sim);
  k_cpb      <<<dim3(2048), dim3(256), 0, stream>>>(vg, cw0, cb0, cw1, cb1, cw2, cb2, sim);
  k_softmax  <<<dim3(2048), dim3(256), 0, stream>>>(sim, attn);
  k_out      <<<dim3(256),  dim3(128), 0, stream>>>(attn, vB, outT);
  k_final    <<<dim3(2048), dim3(128), 0, stream>>>(x1, outT, wo, bo, (float*)d_out);
}
